// HyperGraph_Model_3427383902832
// MI455X (gfx1250) — matrix operand order repaired and checked by reading, not compiled, not run on hardware
//
#include <hip/hip_runtime.h>
#include <hip/hip_bf16.h>
#include <math.h>

// Problem constants (match reference)
#define B_   64
#define NE_  256
#define NN_  1024
#define L_   512
#define HID_ 128
#define NEGC (-9.0e15f)
#define ALPHA_ 0.2f

typedef __bf16 bf16;
typedef __attribute__((ext_vector_type(16))) __bf16 v16bf;
typedef __attribute__((ext_vector_type(8)))  float  v8f;

// ---------------- small device helpers ----------------
__device__ __forceinline__ bf16 f2bf(float f) {
  unsigned u = __float_as_uint(f);
  u += 0x7FFFu + ((u >> 16) & 1u);          // round-to-nearest-even
  unsigned short h = (unsigned short)(u >> 16);
  return __builtin_bit_cast(bf16, h);
}

__device__ __forceinline__ float wave_max(float v) {
#pragma unroll
  for (int off = 16; off > 0; off >>= 1) v = fmaxf(v, __shfl_xor(v, off, 32));
  return v;
}
__device__ __forceinline__ float wave_sum(float v) {
#pragma unroll
  for (int off = 16; off > 0; off >>= 1) v += __shfl_xor(v, off, 32);
  return v;
}

// A fragment (16xK tile, bf16, row-major in LDS with given stride).
// layout used here (contiguous k, same as B): M = lane%16; lanes 0-15 hold K 0-15, lanes 16-31 hold K 16-31.
__device__ __forceinline__ void load_a_frag(const bf16* lds, int kBase, int stride, v16bf& a) {
  const int lane = threadIdx.x & 31;
  const int row  = lane & 15;
  const int half = lane >> 4;
#pragma unroll
  for (int v = 0; v < 8; ++v) {
    const int k0  = kBase + half * 16 + v * 2;
    a[2 * v]     = lds[row * stride + k0];
    a[2 * v + 1] = lds[row * stride + k0 + 1];
  }
}

// B fragment from a K-major bf16 matrix: gT[col*ldk + k].
// layout used here (contiguous k, same as A): N = lane%16; lanes 0-15 hold K 0-15, lanes 16-31 hold K 16-31
// => one 32-byte contiguous vector load per lane, no conversion.
__device__ __forceinline__ v16bf load_b_frag(const bf16* gT, int kBase, int ldk, int colBase) {
  const int lane = threadIdx.x & 31;
  const int col  = colBase + (lane & 15);
  const int half = lane >> 4;
  return *(const v16bf*)(gT + (size_t)col * ldk + kBase + half * 16);
}

#define WMMA_BF16(c, a, b) \
  __builtin_amdgcn_wmma_f32_16x16x32_bf16(false, (a), false, (b), (short)0, (c), false, false)

// ---------------- kernel 0: tiled transpose + f32->bf16 convert --------------
// in : (z, R, 128) f32 row-major    out : (z, 128, R) bf16 (K-major for WMMA B)
__global__ __launch_bounds__(256) void t_conv(const float* __restrict__ in,
                                              bf16* __restrict__ out, int R) {
  __shared__ bf16 t[32][33];
  const float* ip = in  + (size_t)blockIdx.z * R * HID_;
  bf16*        op = out + (size_t)blockIdx.z * R * HID_;
  const int r0 = blockIdx.x * 32, c0 = blockIdx.y * 32;
  const int tx = threadIdx.x & 31, ty = threadIdx.x >> 5;       // 32 x 8
#pragma unroll
  for (int i = 0; i < 32; i += 8)
    t[ty + i][tx] = f2bf(ip[(size_t)(r0 + ty + i) * HID_ + c0 + tx]);
  __syncthreads();
#pragma unroll
  for (int i = 0; i < 32; i += 8)
    op[(size_t)(c0 + ty + i) * R + r0 + tx] = t[tx][ty + i];
}

// ---------------- kernel 1: embedding gather ----------------
__global__ __launch_bounds__(HID_) void gather_emb(const int* __restrict__ items,
                                                   const float* __restrict__ emb,
                                                   float* __restrict__ out) {
  const size_t row = blockIdx.x;                      // B*NN rows
  const int it = items[row];
  out[row * HID_ + threadIdx.x] = emb[(size_t)it * HID_ + threadIdx.x];
}

// ------- kernel 2: Y(R x128) = X(R x128) @ W(128x128), W pre-transposed bf16 -
__global__ __launch_bounds__(256) void gemm128(const float* __restrict__ X,
                                               const bf16* __restrict__ Wt,
                                               float* __restrict__ Y) {
  __shared__ bf16 sA[16 * HID_];                       // 4 KB
  const int rowBase = blockIdx.x * 16;
  for (int i = threadIdx.x; i < 16 * HID_; i += 256) {
    const int r = i >> 7, k = i & 127;
    sA[i] = f2bf(X[(size_t)(rowBase + r) * HID_ + k]);
  }
  __syncthreads();

  const int wave = threadIdx.x >> 5;
  const int colBase = wave * 16;                       // 8 waves -> 128 cols
  v8f c = {};
#pragma unroll
  for (int kb = 0; kb < HID_; kb += 32) {
    v16bf a;
    load_a_frag(sA, kb, HID_, a);
    v16bf b = load_b_frag(Wt, kb, HID_, colBase);
    c = WMMA_BF16(c, a, b);
  }
  const int lane = threadIdx.x & 31;
  const int col  = colBase + (lane & 15);
  const int half = lane >> 4;
#pragma unroll
  for (int v = 0; v < 8; ++v) {
    const int m = v + 8 * half;
    Y[(size_t)(rowBase + m) * HID_ + col] = c[v];
  }
}

// ---------------- kernel 3: out[r] = (leaky?)(dot(M[r,:],vec) + dot(bu,bv)) ---
__global__ __launch_bounds__(256) void row_dot(const float* __restrict__ M,
                                               const float* __restrict__ vec,
                                               const float* __restrict__ bu,
                                               const float* __restrict__ bv,
                                               float* __restrict__ out, int leaky) {
  const int wave = threadIdx.x >> 5, lane = threadIdx.x & 31;
  const size_t row = (size_t)blockIdx.x * 8 + wave;
  const float* m = M + row * HID_;
  float s = 0.f;
#pragma unroll
  for (int k = lane; k < HID_; k += 32) s += m[k] * vec[k];
  s = wave_sum(s);
  float bias = 0.f;
  if (bu) {
    float t = 0.f;
#pragma unroll
    for (int k = lane; k < HID_; k += 32) t += bu[k] * bv[k];
    bias = wave_sum(t);
  }
  if (lane == 0) {
    float e = s + bias;
    if (leaky) e = (e >= 0.f) ? e : ALPHA_ * e;
    out[row] = e;
  }
}

// ------- kernel 4: fused masked softmax over NN + edge = att_edge @ X -------
// one block = one batch, one 16-edge tile; values pre-transposed bf16 (128 x NN)
__global__ __launch_bounds__(256) void edge_aggr(const float* __restrict__ adj,
                                                 const float* __restrict__ e1,
                                                 const bf16* __restrict__ XvT,
                                                 float* __restrict__ Eout) {
  __shared__ float s_e1[NN_];                          // 4 KB
  __shared__ bf16  s_w[16 * NN_];                      // 32 KB exp-weights
  __shared__ float s_sum[16];
  const int tpb  = NE_ / 16;
  const int bIdx = blockIdx.x / tpb;
  const int et   = blockIdx.x % tpb;

  for (int i = threadIdx.x; i < NN_; i += 256) s_e1[i] = e1[(size_t)bIdx * NN_ + i];
  __syncthreads();

  const float* adjB = adj + ((size_t)bIdx * NE_ + (size_t)et * 16) * NN_;
  const int wave = threadIdx.x >> 5, lane = threadIdx.x & 31;
#pragma unroll
  for (int er = 0; er < 2; ++er) {                     // 8 waves * 2 = 16 edges
    const int e = wave * 2 + er;
    const float* arow = adjB + (size_t)e * NN_;
    float mx = -3.402823466e38f;
    for (int n = lane; n < NN_; n += 32) {
      __builtin_prefetch(arow + n + 512, 0, 0);        // global_prefetch_b8
      const float sc = (arow[n] > 0.f) ? s_e1[n] : NEGC;
      mx = fmaxf(mx, sc);
    }
    mx = wave_max(mx);
    float sum = 0.f;
    for (int n = lane; n < NN_; n += 32) {
      const float sc = (arow[n] > 0.f) ? s_e1[n] : NEGC;
      const float w  = __expf(sc - mx);
      s_w[e * NN_ + n] = f2bf(w);
      sum += w;
    }
    sum = wave_sum(sum);
    if (lane == 0) s_sum[e] = sum;
  }
  __syncthreads();

  // WMMA: A = exp-weights (16 x NN) from LDS, B = XvT[b] (K-major); 1/sum at store
  const bf16* xb = XvT + (size_t)bIdx * NN_ * HID_;
  const int colBase = wave * 16;
  v8f c = {};
  for (int kb = 0; kb < NN_; kb += 32) {
    v16bf a;
    load_a_frag(s_w, kb, NN_, a);
    v16bf b = load_b_frag(xb, kb, NN_, colBase);
    c = WMMA_BF16(c, a, b);
  }
  const int col  = colBase + (lane & 15);
  const int half = lane >> 4;
  float* outp = Eout + ((size_t)bIdx * NE_ + (size_t)et * 16) * HID_;
#pragma unroll
  for (int v = 0; v < 8; ++v) {
    const int m = v + 8 * half;
    outp[(size_t)m * HID_ + col] = c[v] * (1.0f / s_sum[m]);
  }
}

// ------- kernel 5: fused masked softmax over NE + node = att_node @ edge ----
// one block = one batch, one 16-node tile; edge values pre-transposed bf16 (128 x NE)
__global__ __launch_bounds__(256) void node_aggr(const float* __restrict__ adj,
                                                 const float* __restrict__ xa,
                                                 const float* __restrict__ ea,
                                                 const bf16* __restrict__ EvT,
                                                 float* __restrict__ Xout, int doElu) {
  __shared__ float s_adj[NE_ * 16];                    // 16 KB column slab
  __shared__ float s_ea[NE_];                          // 1 KB
  __shared__ bf16  s_w[16 * NE_];                      // 8 KB exp-weights
  __shared__ float s_sum[16];
  const int tpb   = NN_ / 16;
  const int bIdx  = blockIdx.x / tpb;
  const int nt    = blockIdx.x % tpb;
  const int nBase = nt * 16;

  const float* adjB = adj + (size_t)bIdx * NE_ * NN_ + nBase;
  for (int i = threadIdx.x; i < NE_ * 16; i += 256) {
    const int e = i >> 4, r = i & 15;
    s_adj[e * 16 + r] = adjB[(size_t)e * NN_ + r];
  }
  for (int i = threadIdx.x; i < NE_; i += 256) s_ea[i] = ea[(size_t)bIdx * NE_ + i];
  __syncthreads();

  const int wave = threadIdx.x >> 5, lane = threadIdx.x & 31;
#pragma unroll
  for (int nr = 0; nr < 2; ++nr) {                     // 8 waves * 2 = 16 nodes
    const int r = wave * 2 + nr;
    const float xar = xa[(size_t)bIdx * NN_ + nBase + r];
    float mx = -3.402823466e38f;
    for (int e = lane; e < NE_; e += 32) {
      float sc;
      if (s_adj[e * 16 + r] > 0.f) {
        const float z = xar + s_ea[e];
        sc = (z >= 0.f) ? z : ALPHA_ * z;
      } else sc = NEGC;
      mx = fmaxf(mx, sc);
    }
    mx = wave_max(mx);
    float sum = 0.f;
    for (int e = lane; e < NE_; e += 32) {
      float sc;
      if (s_adj[e * 16 + r] > 0.f) {
        const float z = xar + s_ea[e];
        sc = (z >= 0.f) ? z : ALPHA_ * z;
      } else sc = NEGC;
      const float w = __expf(sc - mx);
      s_w[r * NE_ + e] = f2bf(w);
      sum += w;
    }
    sum = wave_sum(sum);
    if (lane == 0) s_sum[r] = sum;
  }
  __syncthreads();

  // WMMA: A = exp-weights (16 x NE) from LDS, B = EvT[b] (K-major)
  const bf16* eb = EvT + (size_t)bIdx * NE_ * HID_;
  const int colBase = wave * 16;
  v8f c = {};
#pragma unroll
  for (int kb = 0; kb < NE_; kb += 32) {
    v16bf a;
    load_a_frag(s_w, kb, NE_, a);
    v16bf b = load_b_frag(eb, kb, NE_, colBase);
    c = WMMA_BF16(c, a, b);
  }
  const int col  = colBase + (lane & 15);
  const int half = lane >> 4;
  float* outp = Xout + ((size_t)bIdx * NN_ + nBase) * HID_;
#pragma unroll
  for (int v = 0; v < 8; ++v) {
    const int m = v + 8 * half;
    float val = c[v] * (1.0f / s_sum[m]);
    if (doElu) val = (val > 0.f) ? val : expm1f(val);
    outp[(size_t)m * HID_ + col] = val;
  }
}

// ---------------- kernel 6: gather by alias, masked mean, LayerNorm ----------
__global__ __launch_bounds__(HID_) void readout_k(const float* __restrict__ X,
                                                  const int* __restrict__ alias,
                                                  const float* __restrict__ mask,
                                                  const float* __restrict__ g,
                                                  const float* __restrict__ beta,
                                                  float* __restrict__ out) {
  const int b = blockIdx.x, t = threadIdx.x;
  const int* al = alias + (size_t)b * L_;
  const float* mk = mask + (size_t)b * L_;
  float acc = 0.f, dn = 0.f;
  for (int l = 0; l < L_; ++l) {
    const float m = mk[l];
    acc += X[((size_t)b * NN_ + al[l]) * HID_ + t] * (m * m);
    dn += m;
  }
  const float bv = acc / dn;

  __shared__ float sh[HID_];
  sh[t] = bv; __syncthreads();
  for (int s = HID_ / 2; s > 0; s >>= 1) { if (t < s) sh[t] += sh[t + s]; __syncthreads(); }
  const float mu = sh[0] / (float)HID_; __syncthreads();
  const float d = bv - mu;
  sh[t] = d * d; __syncthreads();
  for (int s = HID_ / 2; s > 0; s >>= 1) { if (t < s) sh[t] += sh[t + s]; __syncthreads(); }
  const float var = sh[0] / (float)HID_;
  out[(size_t)b * HID_ + t] = d * rsqrtf(var + 1e-6f) * g[t] + beta[t];
}

// ---------------- host-side orchestration ----------------
extern "C" void kernel_launch(void* const* d_in, const int* in_sizes, int n_in,
                              void* d_out, int out_size, void* d_ws, size_t ws_size,
                              hipStream_t stream) {
  (void)in_sizes; (void)n_in; (void)out_size; (void)ws_size;
  const int*   items = (const int*)  d_in[0];
  const int*   alias = (const int*)  d_in[1];
  const float* masks = (const float*)d_in[2];
  const float* adj   = (const float*)d_in[3];
  const float* emb   = (const float*)d_in[4];
  const float* w2_1  = (const float*)d_in[5];
  const float* w3_1  = (const float*)d_in[6];
  const float* a_1   = (const float*)d_in[7];
  const float* a2_1  = (const float*)d_in[8];
  const float* wc_1  = (const float*)d_in[9];
  const float* w_2   = (const float*)d_in[10];
  const float* w2_2  = (const float*)d_in[11];
  const float* w3_2  = (const float*)d_in[12];
  const float* a_2   = (const float*)d_in[13];
  const float* a2_2  = (const float*)d_in[14];
  const float* wc_2  = (const float*)d_in[15];
  const float* ln_g  = (const float*)d_in[16];
  const float* ln_b  = (const float*)d_in[17];
  float* out = (float*)d_out;

  // workspace carve-up (256B-aligned slabs)
  char* p = (char*)d_ws;
  auto alloc = [&](size_t bytes) -> void* {
    void* r = (void*)p;
    p += (bytes + 255) & ~(size_t)255;
    return r;
  };
  const size_t NXF = (size_t)B_ * NN_ * HID_;  // node-tensor elements
  const size_t NEF = (size_t)B_ * NE_ * HID_;  // edge-tensor elements
  float* hid   = (float*)alloc(NXF * 4);       // x (L1) -> x1 (L1 node output)
  float* x4    = (float*)alloc(NXF * 4);       // x @ W2 per layer
  float* xalt  = (float*)alloc(NXF * 4);       // x1 @ W (L2) -> x2 (L2 node output)
  float* edge  = (float*)alloc(NEF * 4);
  float* edge4 = (float*)alloc(NEF * 4);
  float* e1    = (float*)alloc((size_t)B_ * NN_ * 4);
  float* xa2   = (float*)alloc((size_t)B_ * NN_ * 4);
  float* ea2   = (float*)alloc((size_t)B_ * NE_ * 4);
  bf16* valT   = (bf16*)alloc(NXF * 2);        // bf16 K-major node values (reused L1/L2)
  bf16* edgeT  = (bf16*)alloc(NEF * 2);        // bf16 K-major edge values (reused L1/L2)
  bf16* wt2_1  = (bf16*)alloc(HID_ * HID_ * 2);
  bf16* wt3_1  = (bf16*)alloc(HID_ * HID_ * 2);
  bf16* wt_2   = (bf16*)alloc(HID_ * HID_ * 2);
  bf16* wt2_2  = (bf16*)alloc(HID_ * HID_ * 2);
  bf16* wt3_2  = (bf16*)alloc(HID_ * HID_ * 2);

  const dim3 blk256(256), blk128(HID_);
  const dim3 gRowsX(B_ * NN_ / 16), gRowsE(B_ * NE_ / 16);
  const dim3 gDotX(B_ * NN_ / 8),  gDotE(B_ * NE_ / 8);
  const dim3 gEdge(B_ * (NE_ / 16)), gNode(B_ * (NN_ / 16));
  const dim3 gTW(HID_ / 32, HID_ / 32, 1);          // weight transpose
  const dim3 gTX(NN_ / 32, HID_ / 32, B_);          // node-values transpose
  const dim3 gTE(NE_ / 32, HID_ / 32, B_);          // edge-values transpose

  // one-time: weights -> bf16 K-major
  t_conv<<<gTW, blk256, 0, stream>>>(w2_1, wt2_1, HID_);
  t_conv<<<gTW, blk256, 0, stream>>>(w3_1, wt3_1, HID_);
  t_conv<<<gTW, blk256, 0, stream>>>(w_2,  wt_2,  HID_);
  t_conv<<<gTW, blk256, 0, stream>>>(w2_2, wt2_2, HID_);
  t_conv<<<gTW, blk256, 0, stream>>>(w3_2, wt3_2, HID_);

  gather_emb<<<dim3(B_ * NN_), blk128, 0, stream>>>(items, emb, hid);

  // -------- layer 1 (transfer=False, concat=True) --------
  gemm128<<<gRowsX, blk256, 0, stream>>>(hid, wt2_1, x4);
  t_conv<<<gTX, blk256, 0, stream>>>(hid, valT, NN_);               // values (x)
  row_dot<<<gDotX, blk256, 0, stream>>>(x4, a_1 + HID_, wc_1, a_1, e1, 1);
  edge_aggr<<<gEdge, blk256, 0, stream>>>(adj, e1, valT, edge);
  gemm128<<<gRowsE, blk256, 0, stream>>>(edge, wt3_1, edge4);
  t_conv<<<gTE, blk256, 0, stream>>>(edge, edgeT, NE_);             // values (edge)
  row_dot<<<gDotX, blk256, 0, stream>>>(x4, a2_1, nullptr, nullptr, xa2, 0);
  row_dot<<<gDotE, blk256, 0, stream>>>(edge4, a2_1 + HID_, nullptr, nullptr, ea2, 0);
  node_aggr<<<gNode, blk256, 0, stream>>>(adj, xa2, ea2, edgeT, hid, /*elu=*/1);

  // -------- layer 2 (transfer=True, concat=False) --------
  gemm128<<<gRowsX, blk256, 0, stream>>>(hid, wt2_2, x4);
  gemm128<<<gRowsX, blk256, 0, stream>>>(hid, wt_2, xalt);          // transfer
  t_conv<<<gTX, blk256, 0, stream>>>(xalt, valT, NN_);              // values (x@W)
  row_dot<<<gDotX, blk256, 0, stream>>>(x4, a_2 + HID_, wc_2, a_2, e1, 1);
  edge_aggr<<<gEdge, blk256, 0, stream>>>(adj, e1, valT, edge);
  gemm128<<<gRowsE, blk256, 0, stream>>>(edge, wt3_2, edge4);
  t_conv<<<gTE, blk256, 0, stream>>>(edge, edgeT, NE_);             // values (edge)
  row_dot<<<gDotX, blk256, 0, stream>>>(x4, a2_2, nullptr, nullptr, xa2, 0);
  row_dot<<<gDotE, blk256, 0, stream>>>(edge4, a2_2 + HID_, nullptr, nullptr, ea2, 0);
  node_aggr<<<gNode, blk256, 0, stream>>>(adj, xa2, ea2, edgeT, xalt, /*elu=*/0);

  // -------- readout --------
  readout_k<<<dim3(B_), blk128, 0, stream>>>(xalt, alias, masks, ln_g, ln_b, out);
}
